// TransformerBlock_40810779246813
// MI455X (gfx1250) — matrix operand order repaired and checked by reading, not compiled, not run on hardware
//
#include <hip/hip_runtime.h>

// ---------------------------------------------------------------------------
// CDNA5 / gfx1250 transformer block: LN -> QKV GEMM -> flash attention ->
// out-proj (+resid) -> LN -> MLP1 (+GELU) -> MLP2 (+resid).
// All matmuls via v_wmma_f32_16x16x32_bf16 (wave32 WMMA), f32 accumulate.
// GEMM K-loop is double-buffered through LDS (global->regs overlap compute).
// Attention computes S^T = K*Q^T and O^T = V^T*P^T so the softmax axis lies
// along VGPRs (1 shfl_xor(16) per query) and P/O stores are b128-vectorized.
// ---------------------------------------------------------------------------

typedef __attribute__((ext_vector_type(16))) __bf16 v16bf;
typedef __attribute__((ext_vector_type(8)))  float  v8f;

union Frag {
    v16bf v;
    uint4 u[2];
};

__device__ inline unsigned short f2bf(float x) {
    union { float f; unsigned int u; } cv;
    cv.f = x;
    unsigned int r = cv.u + 0x7FFFu + ((cv.u >> 16) & 1u);   // round-to-nearest-even
    return (unsigned short)(r >> 16);
}

__device__ inline v8f v8f_zero() {
    v8f z;
#pragma unroll
    for (int i = 0; i < 8; ++i) z[i] = 0.f;
    return z;
}

__device__ inline v8f wmma_bf16(const Frag& a, const Frag& b, v8f c) {
    return __builtin_amdgcn_wmma_f32_16x16x32_bf16(
        false, a.v, false, b.v, (short)0, c, false, false);
}

// Pack 8 scaled floats into 8 bf16 inside a uint4 (memory order v[0]..v[7]).
__device__ inline uint4 pack8_bf16(const v8f& v, float scale) {
    uint4 r;
    r.x = (unsigned)f2bf(v[0] * scale) | ((unsigned)f2bf(v[1] * scale) << 16);
    r.y = (unsigned)f2bf(v[2] * scale) | ((unsigned)f2bf(v[3] * scale) << 16);
    r.z = (unsigned)f2bf(v[4] * scale) | ((unsigned)f2bf(v[5] * scale) << 16);
    r.w = (unsigned)f2bf(v[6] * scale) | ((unsigned)f2bf(v[7] * scale) << 16);
    return r;
}

// ---------------------------------------------------------------------------
// Weight convert: w [K][N] f32  ->  wt [N][K] bf16 (transposed for B-operand)
// ---------------------------------------------------------------------------
__global__ __launch_bounds__(256)
void convert_wt(const float* __restrict__ w, unsigned short* __restrict__ wt,
                int K, int N) {
    long long idx = (long long)blockIdx.x * 256 + threadIdx.x;
    long long total = (long long)K * N;
    if (idx < total) {
        int n = (int)(idx / K);
        int k = (int)(idx % K);
        wt[idx] = f2bf(w[(long long)k * N + n]);
    }
}

// ---------------------------------------------------------------------------
// LayerNorm (one row per block) -> bf16 output
// ---------------------------------------------------------------------------
__global__ __launch_bounds__(256)
void layernorm_to_bf16(const float* __restrict__ x, const float* __restrict__ g,
                       const float* __restrict__ beta,
                       unsigned short* __restrict__ out, int C) {
    __shared__ float red[256];
    const int row = blockIdx.x;
    const int t = threadIdx.x;
    const float* xr = x + (long long)row * C;

    float s = 0.f;
    for (int i = t; i < C; i += 256) s += xr[i];
    red[t] = s;
    __syncthreads();
    for (int st = 128; st > 0; st >>= 1) {
        if (t < st) red[t] += red[t + st];
        __syncthreads();
    }
    const float mu = red[0] * (1.f / C);
    __syncthreads();

    float vs = 0.f;
    for (int i = t; i < C; i += 256) { float d = xr[i] - mu; vs += d * d; }
    red[t] = vs;
    __syncthreads();
    for (int st = 128; st > 0; st >>= 1) {
        if (t < st) red[t] += red[t + st];
        __syncthreads();
    }
    const float rs = rsqrtf(red[0] * (1.f / C) + 1e-5f);

    unsigned short* orow = out + (long long)row * C;
    for (int i = t; i < C; i += 256)
        orow[i] = f2bf((xr[i] - mu) * rs * g[i] + beta[i]);
}

// ---------------------------------------------------------------------------
// Tiled bf16 WMMA GEMM:  out[M][N] = A[M][K] @ Bt[N][K]^T  (+bias,+resid,GELU)
// Block tile 128x256, BK=32, 8 waves (2x4), 64x64 per wave.
// Double-buffered LDS staging: one barrier per K-step, global loads for the
// next tile overlap the 16-WMMA compute of the current tile.
// ---------------------------------------------------------------------------
#define GBM 128
#define GBN 256
#define GBK 32
#define GLD 40   // padded LDS stride in bf16 elements (80B -> 16B aligned rows)

template <bool BIAS, bool RESID, bool GELU>
__global__ __launch_bounds__(256)
void gemm_bf16_wmma(const unsigned short* __restrict__ A,
                    const unsigned short* __restrict__ Bt,
                    const float* __restrict__ bias,
                    const float* __restrict__ resid,
                    float* __restrict__ outF,
                    unsigned short* __restrict__ outB,
                    int M, int N, int K) {
    __shared__ unsigned short sA[2][GBM * GLD];
    __shared__ unsigned short sB[2][GBN * GLD];

    const int tid  = threadIdx.x;
    const int wave = tid >> 5;
    const int lane = tid & 31;
    const int lo   = lane & 15;
    const int hi   = lane >> 4;
    const int wm   = wave >> 2;      // 0..1
    const int wn   = wave & 3;       // 0..3
    const int m0   = blockIdx.y * GBM;
    const int n0   = blockIdx.x * GBN;

    // Per-thread staging chunk coordinates (A: 512 chunks, B: 1024 chunks)
    int aRow[2], aOff[2], bRow[4], bOff[4];
#pragma unroll
    for (int i = 0; i < 2; ++i) {
        int c = tid + i * 256;
        aRow[i] = c >> 2;
        aOff[i] = (c & 3) * 8;
    }
#pragma unroll
    for (int i = 0; i < 4; ++i) {
        int c = tid + i * 256;
        bRow[i] = c >> 2;
        bOff[i] = (c & 3) * 8;
    }

    v8f acc[4][4];
#pragma unroll
    for (int i = 0; i < 4; ++i)
#pragma unroll
        for (int j = 0; j < 4; ++j) acc[i][j] = v8f_zero();

    const int nk = K / GBK;
    uint4 ra[2], rb[4];

    // Prologue: fetch tile 0 and store into buffer 0
#pragma unroll
    for (int i = 0; i < 2; ++i)
        ra[i] = *(const uint4*)(A + (long long)(m0 + aRow[i]) * K + aOff[i]);
#pragma unroll
    for (int i = 0; i < 4; ++i)
        rb[i] = *(const uint4*)(Bt + (long long)(n0 + bRow[i]) * K + bOff[i]);
#pragma unroll
    for (int i = 0; i < 2; ++i)
        *(uint4*)(&sA[0][aRow[i] * GLD + aOff[i]]) = ra[i];
#pragma unroll
    for (int i = 0; i < 4; ++i)
        *(uint4*)(&sB[0][bRow[i] * GLD + bOff[i]]) = rb[i];
    __syncthreads();

    int p = 0;
    for (int kb = 0; kb < nk; ++kb) {
        const bool has_next = (kb + 1 < nk);
        // Issue global loads for the next K-tile (overlap with compute below)
        if (has_next) {
            const int k0 = (kb + 1) * GBK;
#pragma unroll
            for (int i = 0; i < 2; ++i)
                ra[i] = *(const uint4*)(A + (long long)(m0 + aRow[i]) * K + k0 + aOff[i]);
#pragma unroll
            for (int i = 0; i < 4; ++i)
                rb[i] = *(const uint4*)(Bt + (long long)(n0 + bRow[i]) * K + k0 + bOff[i]);
        }

        // Compute on buffer p: keep a[4] resident, stream B fragments
        {
            const unsigned short* cA = sA[p];
            const unsigned short* cB = sB[p];
            Frag a[4];
#pragma unroll
            for (int mt = 0; mt < 4; ++mt) {
                int r = wm * 64 + mt * 16 + lo;
                a[mt].u[0] = *(const uint4*)(&cA[r * GLD + hi * 8]);
                a[mt].u[1] = *(const uint4*)(&cA[r * GLD + 16 + hi * 8]);
            }
#pragma unroll
            for (int nt = 0; nt < 4; ++nt) {
                Frag b;
                int r = wn * 64 + nt * 16 + lo;
                b.u[0] = *(const uint4*)(&cB[r * GLD + hi * 8]);
                b.u[1] = *(const uint4*)(&cB[r * GLD + 16 + hi * 8]);
#pragma unroll
                for (int mt = 0; mt < 4; ++mt)
                    acc[mt][nt] = wmma_bf16(a[mt], b, acc[mt][nt]);
            }
        }

        // Store next tile into the other buffer; single barrier per K-step
        if (has_next) {
            const int q = p ^ 1;
#pragma unroll
            for (int i = 0; i < 2; ++i)
                *(uint4*)(&sA[q][aRow[i] * GLD + aOff[i]]) = ra[i];
#pragma unroll
            for (int i = 0; i < 4; ++i)
                *(uint4*)(&sB[q][bRow[i] * GLD + bOff[i]]) = rb[i];
            __syncthreads();
            p = q;
        }
    }

    // Epilogue: C layout — vgpr r, lane: row = r + hi*8, col = lo
    // (stores are coalesced across the 16 lanes of each half)
#pragma unroll
    for (int mt = 0; mt < 4; ++mt) {
#pragma unroll
        for (int nt = 0; nt < 4; ++nt) {
            int col = n0 + wn * 64 + nt * 16 + lo;
            float bv = BIAS ? bias[col] : 0.f;
#pragma unroll
            for (int r = 0; r < 8; ++r) {
                int row = m0 + wm * 64 + mt * 16 + r + hi * 8;
                long long idx = (long long)row * N + col;
                float v = acc[mt][nt][r];
                if (BIAS) v += bv;
                if (RESID) v += resid[idx];
                if (GELU) v = 0.5f * v * (1.f + erff(v * 0.70710678118654752f));
                if (outF) outF[idx] = v;
                if (outB) outB[idx] = f2bf(v);
            }
        }
    }
}

// ---------------------------------------------------------------------------
// Flash attention (transposed-score formulation).
// qkv bf16 [B*Nseq][3*H*64]; out bf16 [B*Nseq][H*64].
// 256 threads = 8 waves; each wave owns 32 query columns.
//   S^T = K(64xDh) @ Q^T(Dhx32)   -> C layout: col = query (lane), rows = keys
//   softmax axis = keys = VGPR index + hi-half  -> 1 shfl_xor(16) per query
//   O^T = V^T(Dhx64keys) @ P^T(64keysx32)      -> col = query, rows = d
// ---------------------------------------------------------------------------
#define ALD 72   // padded stride (144B rows, 16B aligned)

__global__ __launch_bounds__(256)
void attention_wmma(const unsigned short* __restrict__ qkv,
                    unsigned short* __restrict__ out, int Nseq, int H) {
    __shared__ unsigned short sK[64 * ALD];        // [key][d]
    __shared__ unsigned short sV[64 * ALD];        // transposed: [d][key]
    __shared__ unsigned short sP[8 * 32 * ALD];    // per-wave P: [query][key]

    const int tid  = threadIdx.x;
    const int wave = tid >> 5;
    const int lane = tid & 31;
    const int lo   = lane & 15;
    const int hi   = lane >> 4;
    const int b    = blockIdx.z;
    const int h    = blockIdx.y;
    const int q0   = blockIdx.x * 256 + wave * 32;
    const int LDQ  = 3 * H * 64;                   // 2304
    const long long base = (long long)b * Nseq * LDQ;
    const float SCALE = 0.125f;                    // 1/sqrt(64)

    // Preload Q as B-operand fragments: lane = query column (d chunks
    // {hi*8, 16+hi*8} matching the A-side K fragments).
    Frag qf[2][2];   // [query tile][kstep over d]
#pragma unroll
    for (int nq = 0; nq < 2; ++nq)
#pragma unroll
        for (int ks = 0; ks < 2; ++ks) {
            int n = q0 + nq * 16 + lo;
            const unsigned short* p =
                qkv + base + (long long)n * LDQ + h * 64 + ks * 32 + hi * 8;
            qf[nq][ks].u[0] = *(const uint4*)p;
            qf[nq][ks].u[1] = *(const uint4*)(p + 16);
        }

    v8f o[4][2];                     // O^T tiles: [d tile][query tile]
    float mstate[2], lstate[2];      // per owned query column
#pragma unroll
    for (int dt = 0; dt < 4; ++dt)
#pragma unroll
        for (int nq = 0; nq < 2; ++nq) o[dt][nq] = v8f_zero();
    mstate[0] = mstate[1] = -INFINITY;
    lstate[0] = lstate[1] = 0.f;

    unsigned short* myP = sP + wave * 32 * ALD;

    for (int kt = 0; kt < 16; ++kt) {
        const int key0 = kt * 64;
        __syncthreads();
        // Stage K tile [key][d] (512 x 16B chunks / 256 threads)
#pragma unroll
        for (int i = 0; i < 2; ++i) {
            int c = tid + i * 256;
            int row = c >> 3, kc = c & 7;
            const uint4* src = (const uint4*)(qkv + base +
                (long long)(key0 + row) * LDQ + H * 64 + h * 64 + kc * 8);
            *(uint4*)(&sK[row * ALD + kc * 8]) = *src;
        }
        // Stage V tile transposed -> sV[d][key]
#pragma unroll
        for (int i = 0; i < 2; ++i) {
            int c = tid + i * 256;
            int row = c >> 3, kc = c & 7;
            uint4 d = *(const uint4*)(qkv + base +
                (long long)(key0 + row) * LDQ + 2 * H * 64 + h * 64 + kc * 8);
            const unsigned short* ds = (const unsigned short*)&d;
#pragma unroll
            for (int j = 0; j < 8; ++j)
                sV[(kc * 8 + j) * ALD + row] = ds[j];
        }
        __syncthreads();

        // ---- S^T = K @ Q^T : A = K rows from sK, B = resident Q frags ----
        v8f s[4][2];                 // [key tile][query tile]
#pragma unroll
        for (int mk = 0; mk < 4; ++mk)
#pragma unroll
            for (int nq = 0; nq < 2; ++nq) s[mk][nq] = v8f_zero();
#pragma unroll
        for (int mk = 0; mk < 4; ++mk)
#pragma unroll
            for (int ks = 0; ks < 2; ++ks) {
                Frag kf;  // A layout: lane row = key, d chunks {hi*8, 16+hi*8}
                const unsigned short* p =
                    &sK[(mk * 16 + lo) * ALD + ks * 32 + hi * 8];
                kf.u[0] = *(const uint4*)p;
                kf.u[1] = *(const uint4*)(p + 16);
#pragma unroll
                for (int nq = 0; nq < 2; ++nq)
                    s[mk][nq] = wmma_bf16(kf, qf[nq][ks], s[mk][nq]);
            }

        // ---- online softmax: axis (keys) is local + one cross-half shfl ----
#pragma unroll
        for (int nq = 0; nq < 2; ++nq) {
            // scale scores
#pragma unroll
            for (int mk = 0; mk < 4; ++mk)
#pragma unroll
                for (int r = 0; r < 8; ++r) s[mk][nq][r] *= SCALE;

            float mx = -INFINITY;
#pragma unroll
            for (int mk = 0; mk < 4; ++mk)
#pragma unroll
                for (int r = 0; r < 8; ++r) mx = fmaxf(mx, s[mk][nq][r]);
            mx = fmaxf(mx, __shfl_xor(mx, 16, 32));

            float mold = mstate[nq];
            float mnew = fmaxf(mold, mx);
            float alpha = __expf(mold - mnew);

            float rsum = 0.f;
#pragma unroll
            for (int mk = 0; mk < 4; ++mk)
#pragma unroll
                for (int r = 0; r < 8; ++r) {
                    float p = __expf(s[mk][nq][r] - mnew);
                    s[mk][nq][r] = p;
                    rsum += p;
                }
            rsum += __shfl_xor(rsum, 16, 32);

            lstate[nq] = lstate[nq] * alpha + rsum;
            mstate[nq] = mnew;
#pragma unroll
            for (int dt = 0; dt < 4; ++dt) o[dt][nq] *= alpha;

            // Store P row-major [query][key]: keys for fixed query are the
            // 8 VGPR entries -> one b128 store per key tile.
#pragma unroll
            for (int mk = 0; mk < 4; ++mk) {
                uint4 pk = pack8_bf16(s[mk][nq], 1.f);
                *(uint4*)(&myP[(nq * 16 + lo) * ALD + mk * 16 + hi * 8]) = pk;
            }
        }

        // ---- O^T += V^T @ P^T : A = sV rows (d), B = sP rows (query) ----
#pragma unroll
        for (int ks = 0; ks < 2; ++ks) {
            Frag pf[2];  // B layout: lane col = query, key chunks {hi*8, 16+hi*8}
#pragma unroll
            for (int nq = 0; nq < 2; ++nq) {
                const unsigned short* p =
                    &myP[(nq * 16 + lo) * ALD + ks * 32 + hi * 8];
                pf[nq].u[0] = *(const uint4*)p;
                pf[nq].u[1] = *(const uint4*)(p + 16);
            }
#pragma unroll
            for (int dt = 0; dt < 4; ++dt) {
                Frag vf;  // A layout: lane row = d, key chunks {hi*8, 16+hi*8}
                const unsigned short* pv =
                    &sV[(dt * 16 + lo) * ALD + ks * 32 + hi * 8];
                vf.u[0] = *(const uint4*)pv;
                vf.u[1] = *(const uint4*)(pv + 16);
#pragma unroll
                for (int nq = 0; nq < 2; ++nq)
                    o[dt][nq] = wmma_bf16(vf, pf[nq], o[dt][nq]);
            }
        }
    }

    // Normalize and write O^T: lane col = query, rows d contiguous along r
    // -> one b128 store per (query tile, d tile).
    const int Cout = H * 64;
#pragma unroll
    for (int nq = 0; nq < 2; ++nq) {
        float inv = 1.f / lstate[nq];
        long long row = (long long)b * Nseq + q0 + nq * 16 + lo;
#pragma unroll
        for (int dt = 0; dt < 4; ++dt) {
            uint4 pk = pack8_bf16(o[dt][nq], inv);
            *(uint4*)(&out[row * Cout + h * 64 + dt * 16 + hi * 8]) = pk;
        }
    }
}

// ---------------------------------------------------------------------------
// Host-side orchestration
// ---------------------------------------------------------------------------
extern "C" void kernel_launch(void* const* d_in, const int* in_sizes, int n_in,
                              void* d_out, int out_size, void* d_ws, size_t ws_size,
                              hipStream_t stream) {
    (void)in_sizes; (void)n_in; (void)out_size; (void)ws_size;

    const float* x     = (const float*)d_in[0];
    const float* ln1_g = (const float*)d_in[1];
    const float* ln1_b = (const float*)d_in[2];
    const float* w_qkv = (const float*)d_in[3];
    const float* w_out = (const float*)d_in[4];
    const float* b_out = (const float*)d_in[5];
    const float* ln2_g = (const float*)d_in[6];
    const float* ln2_b = (const float*)d_in[7];
    const float* w1    = (const float*)d_in[8];
    const float* b1    = (const float*)d_in[9];
    const float* w2    = (const float*)d_in[10];
    const float* b2    = (const float*)d_in[11];
    float* out = (float*)d_out;

    const int Bsz = 8, Nseq = 1024, C = 768, H = 12;
    const int M = Bsz * Nseq;      // 8192
    const int NQKV = 3 * C;        // 2304
    const int F = 3072;

    char* ws = (char*)d_ws;
    size_t off = 0;
    auto arena = [&](size_t bytes) -> void* {
        void* p = ws + off;
        off = (off + bytes + 255) & ~(size_t)255;
        return p;
    };
    unsigned short* wqkv_t = (unsigned short*)arena((size_t)NQKV * C * 2);
    unsigned short* wout_t = (unsigned short*)arena((size_t)C * C * 2);
    unsigned short* w1_t   = (unsigned short*)arena((size_t)F * C * 2);
    unsigned short* w2_t   = (unsigned short*)arena((size_t)C * F * 2);
    unsigned short* hbuf   = (unsigned short*)arena((size_t)M * C * 2);   // LN1/LN2 out
    unsigned short* qkvb   = (unsigned short*)arena((size_t)M * NQKV * 2);
    unsigned short* abuf   = (unsigned short*)arena((size_t)M * C * 2);   // attn out
    float*          x1     = (float*)arena((size_t)M * C * 4);
    unsigned short* hmid   = (unsigned short*)arena((size_t)M * F * 2);

    // 1) Weight conversion (f32 [K][N] -> bf16 [N][K])
    convert_wt<<<(C * NQKV) / 256, 256, 0, stream>>>(w_qkv, wqkv_t, C, NQKV);
    convert_wt<<<(C * C)    / 256, 256, 0, stream>>>(w_out, wout_t, C, C);
    convert_wt<<<(C * F)    / 256, 256, 0, stream>>>(w1,    w1_t,   C, F);
    convert_wt<<<(F * C)    / 256, 256, 0, stream>>>(w2,    w2_t,   F, C);

    // 2) LN1
    layernorm_to_bf16<<<M, 256, 0, stream>>>(x, ln1_g, ln1_b, hbuf, C);

    // 3) QKV GEMM -> bf16 qkv
    gemm_bf16_wmma<false, false, false><<<dim3(NQKV / GBN, M / GBM), 256, 0, stream>>>(
        hbuf, wqkv_t, nullptr, nullptr, nullptr, qkvb, M, NQKV, C);

    // 4) Flash attention -> bf16 abuf
    attention_wmma<<<dim3(Nseq / 256, H, Bsz), 256, 0, stream>>>(qkvb, abuf, Nseq, H);

    // 5) Out projection + bias + residual(x) -> f32 x1
    gemm_bf16_wmma<true, true, false><<<dim3(C / GBN, M / GBM), 256, 0, stream>>>(
        abuf, wout_t, b_out, x, x1, nullptr, M, C, C);

    // 6) LN2
    layernorm_to_bf16<<<M, 256, 0, stream>>>(x1, ln2_g, ln2_b, hbuf, C);

    // 7) MLP1 + bias + exact GELU -> bf16 hmid
    gemm_bf16_wmma<true, false, true><<<dim3(F / GBN, M / GBM), 256, 0, stream>>>(
        hbuf, w1_t, b1, nullptr, nullptr, hmid, M, F, C);

    // 8) MLP2 + bias + residual(x1) -> f32 out
    gemm_bf16_wmma<true, true, false><<<dim3(C / GBN, M / GBM), 256, 0, stream>>>(
        hmid, w2_t, b2, x1, out, nullptr, M, C, F);
}
